// MambaEncoderForward_68564857913691
// MI455X (gfx1250) — matrix operand order repaired and checked by reading, not compiled, not run on hardware
//
#include <hip/hip_runtime.h>
#include <cstdint>
#include <cstddef>

// ---------------------------------------------------------------------------
// Problem constants: b=4, L=4096, D=64, dm=128, di=256, ds=16, dconv=4, dtr=8
// ---------------------------------------------------------------------------
#define B_      4
#define LSEQ    4096
#define DMODEL  64
#define DM      128
#define DI      256
#define DS      16
#define DTR     8
#define TTOK    (B_ * LSEQ)      // 16384 tokens
#define LC      256              // scan chunk length
#define NCHUNK  (LSEQ / LC)      // 16 chunks per sequence
#define LOG2E   1.4426950408889634f

typedef __attribute__((ext_vector_type(16))) _Float16 v16h;
typedef __attribute__((ext_vector_type(8)))  _Float16 v8h;
typedef __attribute__((ext_vector_type(8)))  float    v8f;

struct HPair { v8h x, y; };   // 32B -> bit_cast to v16h with no data movement

// ---------------------------------------------------------------------------
// Planar split-f16: value v stored as hi = f16(v) in plane0, lo = f16(v-hi) in
// plane1 (plane1 at +off elements).  A*B ~= AhiBhi + AhiBlo + AloBhi on the
// f16 WMMA path with f32 accumulation gives near-fp32 precision; planar layout
// means fragments are straight 16B loads (no unpack ALU).
// ---------------------------------------------------------------------------
__device__ __forceinline__ void store_split(_Float16* __restrict__ base, size_t idx,
                                            size_t off, float v) {
  _Float16 h = (_Float16)v;
  base[idx]       = h;
  base[idx + off] = (_Float16)(v - (float)h);
}
__device__ __forceinline__ float sigmoidf_(float x) { return 1.f / (1.f + __expf(-x)); }

// A fragment: lane row = lane&15, sel = lane>>4
//   elems j -> K = k0 + 16*sel + j (contiguous 16)
__device__ __forceinline__ v16h ld_fragA(const _Float16* __restrict__ p) {
  HPair h{*(const v8h*)p, *(const v8h*)(p + 8)};
  return __builtin_bit_cast(v16h, h);
}
// B fragment: lane col = lane&15, elems j -> K = k0 + 16*sel + j (contiguous 16)
__device__ __forceinline__ v16h ld_fragB(const _Float16* __restrict__ p) {
  HPair h{*(const v8h*)p, *(const v8h*)(p + 8)};
  return __builtin_bit_cast(v16h, h);
}

#define WMMA3(ACC, AH, AL, BH, BL)                                                              \
  ACC = __builtin_amdgcn_wmma_f32_16x16x32_f16(false, AH, false, BH, (short)0, ACC, false, false); \
  ACC = __builtin_amdgcn_wmma_f32_16x16x32_f16(false, AH, false, BL, (short)0, ACC, false, false); \
  ACC = __builtin_amdgcn_wmma_f32_16x16x32_f16(false, AL, false, BH, (short)0, ACC, false, false);

// ---------------------------------------------------------------------------
// WMMA GEMM: out[M,N] = A[M,K] @ W[N,K]^T (+bias)(+resid)(+act)
// Planar split-f16 operands. One wave computes a 32x32 tile (2x2 16x16
// sub-tiles): 12 WMMAs per K-32 step vs 16 b128 loads. 4 waves/block cover
// 128 consecutive rows. Requires M%128==0, N%32==0, K%32==0 (all true here).
// ACT: 0 = none, 1 = softplus (for delta).
// ---------------------------------------------------------------------------
template <bool HB, bool HR, bool OF, bool OP, int ACT>
__global__ __launch_bounds__(128) void gemm_split(
    const _Float16* __restrict__ A, size_t Aoff,
    const _Float16* __restrict__ W, size_t Woff,
    const float* __restrict__ bias, const float* __restrict__ resid,
    float* __restrict__ outF, _Float16* __restrict__ outP, size_t Poff,
    int N, int K)
{
  const int lane = threadIdx.x & 31;
  const int wv   = threadIdx.x >> 5;
  const int m0   = (blockIdx.y * 4 + wv) * 32;
  const int n0   = blockIdx.x * 32;
  const int rl   = lane & 15;
  const int sel  = lane >> 4;

  const _Float16* Ar0 = A + (size_t)(m0 + rl) * K;
  const _Float16* Ar1 = A + (size_t)(m0 + 16 + rl) * K;
  const _Float16* Wc0 = W + (size_t)(n0 + rl) * K;
  const _Float16* Wc1 = W + (size_t)(n0 + 16 + rl) * K;

  v8f acc[2][2];
#pragma unroll
  for (int mi = 0; mi < 2; ++mi)
#pragma unroll
    for (int ni = 0; ni < 2; ++ni)
#pragma unroll
      for (int r = 0; r < 8; ++r) acc[mi][ni][r] = 0.f;

  for (int k0 = 0; k0 < K; k0 += 32) {
    const int ao = k0 + 16 * sel;
    const int bo = k0 + 16 * sel;
    v16h a0h = ld_fragA(Ar0 + ao);
    v16h a0l = ld_fragA(Ar0 + Aoff + ao);
    v16h a1h = ld_fragA(Ar1 + ao);
    v16h a1l = ld_fragA(Ar1 + Aoff + ao);
    v16h b0h = ld_fragB(Wc0 + bo);
    v16h b0l = ld_fragB(Wc0 + Woff + bo);
    v16h b1h = ld_fragB(Wc1 + bo);
    v16h b1l = ld_fragB(Wc1 + Woff + bo);
    WMMA3(acc[0][0], a0h, a0l, b0h, b0l)
    WMMA3(acc[0][1], a0h, a0l, b1h, b1l)
    WMMA3(acc[1][0], a1h, a1l, b0h, b0l)
    WMMA3(acc[1][1], a1h, a1l, b1h, b1l)
  }

#pragma unroll
  for (int mi = 0; mi < 2; ++mi) {
#pragma unroll
    for (int ni = 0; ni < 2; ++ni) {
      const int ccol = n0 + ni * 16 + rl;
      float bs = 0.f;
      if (HB) bs = bias[ccol];
#pragma unroll
      for (int r = 0; r < 8; ++r) {
        const int crow = m0 + mi * 16 + sel * 8 + r;
        float v = acc[mi][ni][r] + bs;
        if (ACT == 1) v = (v > 20.f) ? v : log1pf(__expf(v));  // softplus
        const size_t idx = (size_t)crow * N + ccol;
        if (HR) v += resid[idx];
        if (OF) outF[idx] = v;
        if (OP) store_split(outP, idx, Poff, v);
      }
    }
  }
}

// ---------------------------------------------------------------------------
// Prep kernels
// ---------------------------------------------------------------------------
// pack n_tot elements; sources beyond n_src are zero (row-prefix padding)
__global__ void pack_pad(const float* __restrict__ s, _Float16* __restrict__ d,
                         int n_src, int n_tot, size_t off) {
  int i = blockIdx.x * blockDim.x + threadIdx.x;
  if (i >= n_tot) return;
  store_split(d, i, off, (i < n_src) ? s[i] : 0.f);
}
// pack (rows, ksrc) into (rows, kdst) with zero K-padding (for dt_proj_w 8->32)
__global__ void pack_pad_rows(const float* __restrict__ s, _Float16* __restrict__ d,
                              int rows, int ksrc, int kdst, size_t off) {
  int i = blockIdx.x * blockDim.x + threadIdx.x;
  if (i >= rows * kdst) return;
  const int r = i / kdst, c = i - r * kdst;
  store_split(d, i, off, (c < ksrc) ? s[r * ksrc + c] : 0.f);
}
// A = -exp(A_log) * log2(e)  (so the scan uses a bare exp2)
__global__ void negexp2_k(const float* __restrict__ s, float* __restrict__ d, int n) {
  int i = blockIdx.x * blockDim.x + threadIdx.x;
  if (i < n) d[i] = -expf(s[i]) * LOG2E;
}

// RMSNorm over rows of 64 (wave32: 2 elems/lane) -> planar split-f16
__global__ __launch_bounds__(256) void rmsnorm_pack_d64(
    const float* __restrict__ X, const float* __restrict__ w,
    _Float16* __restrict__ outP, size_t off)
{
  const int lane = threadIdx.x & 31;
  const int row  = blockIdx.x * 8 + (threadIdx.x >> 5);
  const float x0 = X[(size_t)row * 64 + lane];
  const float x1 = X[(size_t)row * 64 + lane + 32];
  float ss = x0 * x0 + x1 * x1;
#pragma unroll
  for (int m = 16; m >= 1; m >>= 1) ss += __shfl_xor(ss, m, 32);
  const float sc = rsqrtf(ss * (1.f / 64.f) + 1e-5f);
  store_split(outP, (size_t)row * 64 + lane,      off, x0 * sc * w[lane]);
  store_split(outP, (size_t)row * 64 + lane + 32, off, x1 * sc * w[lane + 32]);
}

// m2 = rmsnorm(M,128) * w + U    (PostNormMamba residual)
__global__ __launch_bounds__(256) void rmsnorm_add_d128(
    const float* __restrict__ M, const float* __restrict__ w,
    const float* __restrict__ U, float* __restrict__ out)
{
  const int lane = threadIdx.x & 31;
  const int row  = blockIdx.x * 8 + (threadIdx.x >> 5);
  float v[4];
  float ss = 0.f;
#pragma unroll
  for (int q = 0; q < 4; ++q) {
    v[q] = M[(size_t)row * 128 + lane + 32 * q];
    ss += v[q] * v[q];
  }
#pragma unroll
  for (int m = 16; m >= 1; m >>= 1) ss += __shfl_xor(ss, m, 32);
  const float sc = rsqrtf(ss * (1.f / 128.f) + 1e-5f);
#pragma unroll
  for (int q = 0; q < 4; ++q) {
    const int j = lane + 32 * q;
    out[(size_t)row * 128 + j] = v[q] * sc * w[j] + U[(size_t)row * 128 + j];
  }
}

// Causal depthwise conv (dconv=4) + bias + SiLU; writes f32 (for scan) and
// planar split-f16 (for the x_proj WMMA GEMM).
__global__ void dwconv_silu(const float* __restrict__ xz, const float* __restrict__ cw,
                            const float* __restrict__ cb, float* __restrict__ xc,
                            _Float16* __restrict__ xcp, size_t off, int total)
{
  int i = blockIdx.x * blockDim.x + threadIdx.x;
  if (i >= total) return;
  const int c = i & 255;
  const int t = i >> 8;
  const int l = t & (LSEQ - 1);
  float acc = cb[c];
#pragma unroll
  for (int k = 0; k < 4; ++k) {
    const int ll = l - 3 + k;
    if (ll >= 0) acc += cw[c * 4 + k] * xz[(size_t)(t - 3 + k) * 512 + c];
  }
  acc = acc * sigmoidf_(acc);   // SiLU
  xc[i] = acc;
  store_split(xcp, i, off, acc);
}

// From x_dbl (T,64 f32; cols 0..7=dt, 8..23=B, 24..39=C, rest zero-pad):
// write B,C (f32) and dt rows K-padded 8->32 as planar split-f16 for the
// dt_proj WMMA GEMM.
__global__ void extract_dtBC(const float* __restrict__ xdbl, _Float16* __restrict__ dtp,
                             size_t off, float* __restrict__ Bm, float* __restrict__ Cm,
                             int n)
{
  int i = blockIdx.x * blockDim.x + threadIdx.x;
  if (i >= n) return;              // n = T*32
  const int c = i & 31;
  const int t = i >> 5;
  store_split(dtp, i, off, (c < DTR) ? xdbl[(size_t)t * 64 + c] : 0.f);
  if (c < DS) {
    Bm[(size_t)t * DS + c] = xdbl[(size_t)t * 64 + DTR + c];
    Cm[(size_t)t * DS + c] = xdbl[(size_t)t * 64 + DTR + DS + c];
  }
}

// ---------------------------------------------------------------------------
// Chunked parallel selective scan: h[l] = exp2(delta*A2)*h[l-1] + (delta*x)*B
// (A2 pre-scaled by log2e).  16384 (b,chunk,d) threads = 512 waves.
// ---------------------------------------------------------------------------
__global__ __launch_bounds__(256) void scan_passA(
    const float* __restrict__ delta, const float* __restrict__ xconv,
    const float* __restrict__ Bm, const float* __restrict__ Amat,
    float* __restrict__ hst, float* __restrict__ Pst)
{
  const int d  = threadIdx.x;
  const int bc = blockIdx.x;
  const int bb = bc >> 4;
  const int ch = bc & (NCHUNK - 1);
  const int t0 = bb * LSEQ + ch * LC;
  float h[DS], P[DS], Ar[DS];
#pragma unroll
  for (int s = 0; s < DS; ++s) { h[s] = 0.f; P[s] = 1.f; Ar[s] = Amat[d * DS + s]; }
  for (int l = 0; l < LC; ++l) {
    const int t = t0 + l;
    const float dl = delta[(size_t)t * DI + d];
    const float xv = xconv[(size_t)t * DI + d];
    const float dx = dl * xv;
    float bv[DS];
    const float4* Bp4 = reinterpret_cast<const float4*>(Bm + (size_t)t * DS);
#pragma unroll
    for (int q = 0; q < 4; ++q) *reinterpret_cast<float4*>(bv + 4 * q) = Bp4[q];
#pragma unroll
    for (int s = 0; s < DS; ++s) {
      const float a = exp2f(dl * Ar[s]);
      h[s] = a * h[s] + dx * bv[s];
      P[s] *= a;
    }
  }
  const size_t o = ((size_t)bc * DI + d) * DS;
#pragma unroll
  for (int s = 0; s < DS; ++s) { hst[o + s] = h[s]; Pst[o + s] = P[s]; }
}

// serial scan across the 16 chunk summaries per (b,d,s) -> per-chunk h0
__global__ void scan_fix(const float* __restrict__ hst, const float* __restrict__ Pst,
                         float* __restrict__ h0st)
{
  const int idx = blockIdx.x * blockDim.x + threadIdx.x;   // 0..B_*DI-1
  const int bb = idx >> 8;
  const int d  = idx & 255;
  float run[DS];
#pragma unroll
  for (int s = 0; s < DS; ++s) run[s] = 0.f;
  for (int c = 0; c < NCHUNK; ++c) {
    const size_t o = (((size_t)(bb * NCHUNK + c)) * DI + d) * DS;
#pragma unroll
    for (int s = 0; s < DS; ++s) {
      h0st[o + s] = run[s];
      run[s] = Pst[o + s] * run[s] + hst[o + s];
    }
  }
}

// replay from true h0; fuse y = sum_s h*C + x*D, SiLU(z) gate, split-f16 pack
__global__ __launch_bounds__(256) void scan_passC(
    const float* __restrict__ delta, const float* __restrict__ xconv,
    const float* __restrict__ Bm, const float* __restrict__ Cm,
    const float* __restrict__ xz, const float* __restrict__ Amat,
    const float* __restrict__ Dp, const float* __restrict__ h0st,
    _Float16* __restrict__ y_pk, size_t yoff)
{
  const int d  = threadIdx.x;
  const int bc = blockIdx.x;
  const int bb = bc >> 4;
  const int ch = bc & (NCHUNK - 1);
  const int t0 = bb * LSEQ + ch * LC;
  float h[DS], Ar[DS];
  const size_t o = ((size_t)bc * DI + d) * DS;
#pragma unroll
  for (int s = 0; s < DS; ++s) { h[s] = h0st[o + s]; Ar[s] = Amat[d * DS + s]; }
  const float Dd = Dp[d];
  for (int l = 0; l < LC; ++l) {
    const int t = t0 + l;
    const float dl = delta[(size_t)t * DI + d];
    const float xv = xconv[(size_t)t * DI + d];
    const float dx = dl * xv;
    float bv[DS], cv[DS];
    const float4* Bp4 = reinterpret_cast<const float4*>(Bm + (size_t)t * DS);
    const float4* Cp4 = reinterpret_cast<const float4*>(Cm + (size_t)t * DS);
#pragma unroll
    for (int q = 0; q < 4; ++q) {
      *reinterpret_cast<float4*>(bv + 4 * q) = Bp4[q];
      *reinterpret_cast<float4*>(cv + 4 * q) = Cp4[q];
    }
    float y = 0.f;
#pragma unroll
    for (int s = 0; s < DS; ++s) {
      const float a = exp2f(dl * Ar[s]);
      h[s] = a * h[s] + dx * bv[s];
      y += h[s] * cv[s];
    }
    y += xv * Dd;
    const float zv = xz[(size_t)t * 512 + DI + d];   // z half of in_proj output
    y *= zv * sigmoidf_(zv);                         // y * silu(z)
    store_split(y_pk, (size_t)t * DI + d, yoff, y);
  }
}

// g = swish(wg) * m2  (planar split-f16 for the final GEMM)
__global__ void gate_pack(const float* __restrict__ wg, const float* __restrict__ m2,
                          _Float16* __restrict__ gp, size_t off, int n)
{
  int i = blockIdx.x * blockDim.x + threadIdx.x;
  if (i >= n) return;
  const float w = wg[i];
  store_split(gp, i, off, w * sigmoidf_(w) * m2[i]);
}

// ---------------------------------------------------------------------------
// Host launcher
// ---------------------------------------------------------------------------
extern "C" void kernel_launch(void* const* d_in, const int* in_sizes, int n_in,
                              void* d_out, int out_size, void* d_ws, size_t ws_size,
                              hipStream_t stream)
{
  (void)in_sizes; (void)n_in; (void)out_size; (void)ws_size;

  const float* x          = (const float*)d_in[0];
  const float* w_norm1    = (const float*)d_in[1];
  const float* w_norm2    = (const float*)d_in[2];
  const float* W_ip       = (const float*)d_in[3];
  const float* b_ip       = (const float*)d_in[4];
  const float* W_fp       = (const float*)d_in[5];
  const float* b_fp       = (const float*)d_in[6];
  const float* W_wp       = (const float*)d_in[7];
  const float* b_wp       = (const float*)d_in[8];
  const float* W_op       = (const float*)d_in[9];
  const float* b_op       = (const float*)d_in[10];
  const float* in_proj_w  = (const float*)d_in[11];
  const float* conv_w     = (const float*)d_in[12];
  const float* conv_b     = (const float*)d_in[13];
  const float* x_proj_w   = (const float*)d_in[14];
  const float* dt_proj_w  = (const float*)d_in[15];
  const float* dt_proj_b  = (const float*)d_in[16];
  const float* A_log      = (const float*)d_in[17];
  const float* D_param    = (const float*)d_in[18];
  const float* mamba_out_w = (const float*)d_in[19];
  float* out = (float*)d_out;

  // bump allocator: alloc(n) returns region of n*4 bytes (256B aligned) =
  // two split-f16 planes of n elements, or n f32.
  size_t off = 0;
  auto alloc = [&](size_t elems) -> void* {
    void* p = (char*)d_ws + off;
    off += (elems * 4 + 255) & ~(size_t)255;
    return p;
  };
  _Float16* wip_pk  = (_Float16*)alloc(DM * DMODEL);        const size_t wip_o  = DM * DMODEL;
  _Float16* wfp_pk  = (_Float16*)alloc(DM * DM);            const size_t wfp_o  = DM * DM;
  _Float16* wwp_pk  = (_Float16*)alloc(DM * DMODEL);        const size_t wwp_o  = DM * DMODEL;
  _Float16* wop_pk  = (_Float16*)alloc(DMODEL * DM);        const size_t wop_o  = DMODEL * DM;
  _Float16* inpj_pk = (_Float16*)alloc(2 * DI * DM);        const size_t inpj_o = 2 * DI * DM;
  _Float16* mout_pk = (_Float16*)alloc(DM * DI);            const size_t mout_o = DM * DI;
  _Float16* xpw_pk  = (_Float16*)alloc(64 * DI);            const size_t xpw_o  = 64 * DI;   // N padded 40->64
  _Float16* dtw_pk  = (_Float16*)alloc(DI * 32);            const size_t dtw_o  = DI * 32;   // K padded 8->32
  float*    Amat    = (float*)   alloc(DI * DS);
  _Float16* xn_pk   = (_Float16*)alloc((size_t)TTOK * DMODEL); const size_t xn_o  = (size_t)TTOK * DMODEL;
  _Float16* ssm1_pk = (_Float16*)alloc((size_t)TTOK * DM);     const size_t ss_o  = (size_t)TTOK * DM;
  float*    u_f     = (float*)   alloc((size_t)TTOK * DM);
  _Float16* u_pk    = (_Float16*)alloc((size_t)TTOK * DM);     const size_t u_o   = (size_t)TTOK * DM;
  float*    xz      = (float*)   alloc((size_t)TTOK * 2 * DI);
  float*    xconv   = (float*)   alloc((size_t)TTOK * DI);
  _Float16* xc_pk   = (_Float16*)alloc((size_t)TTOK * DI);     const size_t xc_o  = (size_t)TTOK * DI;
  float*    xdbl    = (float*)   alloc((size_t)TTOK * 64);
  _Float16* dtp_pk  = (_Float16*)alloc((size_t)TTOK * 32);     const size_t dtp_o = (size_t)TTOK * 32;
  float*    Bm      = (float*)   alloc((size_t)TTOK * DS);
  float*    Cm      = (float*)   alloc((size_t)TTOK * DS);
  float*    delta   = (float*)   alloc((size_t)TTOK * DI);
  float*    hst     = (float*)   alloc((size_t)B_ * NCHUNK * DI * DS);
  float*    Pst     = (float*)   alloc((size_t)B_ * NCHUNK * DI * DS);
  float*    h0st    = (float*)   alloc((size_t)B_ * NCHUNK * DI * DS);
  _Float16* y_pk    = (_Float16*)alloc((size_t)TTOK * DI);     const size_t y_o   = (size_t)TTOK * DI;
  float*    mbuf    = (float*)   alloc((size_t)TTOK * DM);
  float*    m2      = (float*)   alloc((size_t)TTOK * DM);
  float*    wg      = (float*)   alloc((size_t)TTOK * DM);
  _Float16* g_pk    = (_Float16*)alloc((size_t)TTOK * DM);     const size_t g_o   = (size_t)TTOK * DM;

  const dim3 blk256(256), blk128(128);
  auto grid1 = [](int n) { return dim3((unsigned)((n + 255) / 256)); };
  const unsigned GY = TTOK / 128;   // 32-row tiles x 4 waves/block

  // ---- weight prep -------------------------------------------------------
  pack_pad<<<grid1(DM * DMODEL), blk256, 0, stream>>>(W_ip, wip_pk, DM * DMODEL, DM * DMODEL, wip_o);
  pack_pad<<<grid1(DM * DM),     blk256, 0, stream>>>(W_fp, wfp_pk, DM * DM,     DM * DM,     wfp_o);
  pack_pad<<<grid1(DM * DMODEL), blk256, 0, stream>>>(W_wp, wwp_pk, DM * DMODEL, DM * DMODEL, wwp_o);
  pack_pad<<<grid1(DMODEL * DM), blk256, 0, stream>>>(W_op, wop_pk, DMODEL * DM, DMODEL * DM, wop_o);
  pack_pad<<<grid1(2 * DI * DM), blk256, 0, stream>>>(in_proj_w, inpj_pk, 2 * DI * DM, 2 * DI * DM, inpj_o);
  pack_pad<<<grid1(DM * DI),     blk256, 0, stream>>>(mamba_out_w, mout_pk, DM * DI, DM * DI, mout_o);
  pack_pad<<<grid1(64 * DI),     blk256, 0, stream>>>(x_proj_w, xpw_pk, (DTR + 2 * DS) * DI, 64 * DI, xpw_o);
  pack_pad_rows<<<grid1(DI * 32), blk256, 0, stream>>>(dt_proj_w, dtw_pk, DI, DTR, 32, dtw_o);
  negexp2_k<<<grid1(DI * DS), blk256, 0, stream>>>(A_log, Amat, DI * DS);

  // ---- xn = rmsnorm(x, w_norm1) ------------------------------------------
  rmsnorm_pack_d64<<<dim3(TTOK / 8), blk256, 0, stream>>>(x, w_norm1, xn_pk, xn_o);

  // G1: ssm1 = xn @ W_ip^T + b_ip               (T,128) K=64
  gemm_split<true, false, false, true, 0><<<dim3(DM / 32, GY), blk128, 0, stream>>>(
      xn_pk, xn_o, wip_pk, wip_o, b_ip, nullptr, nullptr, ssm1_pk, ss_o, DM, DMODEL);
  // G2: u = ssm1 @ W_fp^T + b_fp                (T,128) K=128  (f32 + packed)
  gemm_split<true, false, true, true, 0><<<dim3(DM / 32, GY), blk128, 0, stream>>>(
      ssm1_pk, ss_o, wfp_pk, wfp_o, b_fp, nullptr, u_f, u_pk, u_o, DM, DM);
  // G3: xz = u @ in_proj_w^T                    (T,512) K=128
  gemm_split<false, false, true, false, 0><<<dim3(2 * DI / 32, GY), blk128, 0, stream>>>(
      u_pk, u_o, inpj_pk, inpj_o, nullptr, nullptr, xz, nullptr, 0, 2 * DI, DM);

  // depthwise causal conv + SiLU
  dwconv_silu<<<grid1(TTOK * DI), blk256, 0, stream>>>(xz, conv_w, conv_b, xconv, xc_pk, xc_o,
                                                       TTOK * DI);

  // Gxp: x_dbl = xconv @ x_proj_w^T (N padded to 64)   (T,64) K=256
  gemm_split<false, false, true, false, 0><<<dim3(64 / 32, GY), blk128, 0, stream>>>(
      xc_pk, xc_o, xpw_pk, xpw_o, nullptr, nullptr, xdbl, nullptr, 0, 64, DI);

  // split dt/B/C; dt K-padded to 32, packed
  extract_dtBC<<<grid1(TTOK * 32), blk256, 0, stream>>>(xdbl, dtp_pk, dtp_o, Bm, Cm, TTOK * 32);

  // Gdt: delta = softplus(dt @ dt_proj_w^T + dt_proj_b)  (T,256) K=32
  gemm_split<true, false, true, false, 1><<<dim3(DI / 32, GY), blk128, 0, stream>>>(
      dtp_pk, dtp_o, dtw_pk, dtw_o, dt_proj_b, nullptr, delta, nullptr, 0, DI, 32);

  // ---- chunked parallel selective scan -----------------------------------
  scan_passA<<<dim3(B_ * NCHUNK), blk256, 0, stream>>>(delta, xconv, Bm, Amat, hst, Pst);
  scan_fix<<<dim3(B_), blk256, 0, stream>>>(hst, Pst, h0st);
  scan_passC<<<dim3(B_ * NCHUNK), blk256, 0, stream>>>(delta, xconv, Bm, Cm, xz, Amat,
                                                       D_param, h0st, y_pk, y_o);

  // G4: m = y @ mamba_out_w^T                   (T,128) K=256
  gemm_split<false, false, true, false, 0><<<dim3(DM / 32, GY), blk128, 0, stream>>>(
      y_pk, y_o, mout_pk, mout_o, nullptr, nullptr, mbuf, nullptr, 0, DM, DI);

  // m2 = rmsnorm(m, w_norm2) + u
  rmsnorm_add_d128<<<dim3(TTOK / 8), blk256, 0, stream>>>(mbuf, w_norm2, u_f, m2);

  // G5: wg = xn @ W_wp^T + b_wp                 (T,128) K=64
  gemm_split<true, false, true, false, 0><<<dim3(DM / 32, GY), blk128, 0, stream>>>(
      xn_pk, xn_o, wwp_pk, wwp_o, b_wp, nullptr, wg, nullptr, 0, DM, DMODEL);

  // g = swish(wg) * m2
  gate_pack<<<grid1(TTOK * DM), blk256, 0, stream>>>(wg, m2, g_pk, g_o, TTOK * DM);

  // G6: out = g @ W_op^T + b_op + x (residual) -> d_out   (T,64) K=128
  gemm_split<true, true, true, false, 0><<<dim3(DMODEL / 32, GY), blk128, 0, stream>>>(
      g_pk, g_o, wop_pk, wop_o, b_op, x, out, nullptr, 0, DMODEL, DM);
}
